// AttentionBlock_28621662060967
// MI455X (gfx1250) — matrix operand order repaired and checked by reading, not compiled, not run on hardware
//
#include <hip/hip_runtime.h>
#include <math.h>

typedef __attribute__((ext_vector_type(16))) _Float16 v16h;
typedef __attribute__((ext_vector_type(8)))  _Float16 v8h;
typedef __attribute__((ext_vector_type(8)))  float    v8f;

namespace {
constexpr int kNB = 32;    // batches
constexpr int kS  = 1024;  // sequence length (H*W)
constexpr int kC  = 256;   // input channels
constexpr int kKS = 128;   // Q/K channels
constexpr int kVS = 256;   // V channels
constexpr float kScale = 0.08838834764831843f;  // 1/sqrt(KS)

// workspace layout (units: _Float16 elements). Total ~33.8 MB (L2-resident).
constexpr size_t WQ_PACK = 0;                               // 8n x 8k tiles * 512
constexpr size_t WK_PACK = WQ_PACK + (size_t)8 * 8 * 512;
constexpr size_t WV_PACK = WK_PACK + (size_t)8 * 8 * 512;   // 16n x 8k tiles
constexpr size_t QH_OFF  = WV_PACK + (size_t)16 * 8 * 512;  // [b][s][ks] f16 (pre-scaled)
constexpr size_t KH_OFF  = QH_OFF + (size_t)kNB * kS * kKS; // [b][s][ks] f16
constexpr size_t VT_OFF  = KH_OFF + (size_t)kNB * kS * kKS; // [b][vs][s] f16 (transposed)
constexpr int kSMS = 1026;  // padded LDS row stride (halves) -> conflict-free
}

__device__ __forceinline__ v8f wmma16x16x32(v16h a, v16h b, v8f c) {
  // D = A(16x32 f16) * B(32x16 f16) + C(16x16 f32)
  return __builtin_amdgcn_wmma_f32_16x16x32_f16(false, a, false, b, (short)0, c,
                                                false, false);
}

// ---------------------------------------------------------------------------
// Kernel 1: pack fp32 weights [C][N] into f16 WMMA B-fragment layout.
// Per 32x16 (KxN) tile: 32 lanes x 16 halves contiguous.
// Lane layout (contiguous k, matching the A-fragments): N = lane&15, K = (lane>>4)*16 + e.
// ---------------------------------------------------------------------------
__global__ void __launch_bounds__(128) pack_weights_kernel(
    const float* __restrict__ Wq, const float* __restrict__ Wk,
    const float* __restrict__ Wv, _Float16* __restrict__ ws) {
  const int tid  = blockIdx.x * blockDim.x + threadIdx.x;
  const int lane = tid & 31;
  const int tile = tid >> 5;  // 0..255 (uniform per wave)
  const float* W;
  _Float16* dst;
  int N, lt;
  if (tile < 64)       { W = Wq; dst = ws + WQ_PACK; N = kKS; lt = tile; }
  else if (tile < 128) { W = Wk; dst = ws + WK_PACK; N = kKS; lt = tile - 64; }
  else                 { W = Wv; dst = ws + WV_PACK; N = kVS; lt = tile - 128; }
  const int ntile = lt >> 3;
  const int kt    = lt & 7;
  const int n  = ntile * 16 + (lane & 15);
  const int k0 = kt * 32 + (lane >> 4) * 16;
  v16h hv;
  #pragma unroll
  for (int e = 0; e < 16; ++e) hv[e] = (_Float16)W[(size_t)(k0 + e) * N + n];
  *(v16h*)(dst + ((size_t)lt * 32 + lane) * 16) = hv;
}

// ---------------------------------------------------------------------------
// Kernel 2: QKV projection. 1 block = (batch, 16-row tile), 4 waves:
//   wave 0 -> Q (scaled by 1/sqrt(KS)), wave 1 -> K, waves 2,3 -> V.
// Output tiles processed in PAIRS: two independent accumulator chains share
// the A-fragments, B-fragments double-buffered across the k-loop, so WMMAs
// interleave instead of lock-stepping on one load->wait->wmma chain.
// V stored transposed [vs][s] so attention B-fragments load contiguously.
// ---------------------------------------------------------------------------
__global__ void __launch_bounds__(128) qkv_proj_kernel(
    const float* __restrict__ x, const float* __restrict__ bq,
    const float* __restrict__ bk, const float* __restrict__ bv,
    _Float16* __restrict__ ws) {
  const int blk   = blockIdx.x;
  const int b     = blk >> 6;
  const int rtile = blk & 63;
  const int wave  = threadIdx.x >> 5;
  const int lane  = threadIdx.x & 31;
  const int lrow  = lane & 15;
  const int lhi   = lane >> 4;
  const int kb8   = lhi * 16;

  const float* xrow = x + (size_t)(b * kS + rtile * 16 + lrow) * kC;
  v16h afrag[8];
  #pragma unroll
  for (int kt = 0; kt < 8; ++kt) {
    v8f c0 = *(const v8f*)(xrow + kt * 32 + kb8);
    v8f c1 = *(const v8f*)(xrow + kt * 32 + kb8 + 8);
    v16h a;
    #pragma unroll
    for (int e = 0; e < 8; ++e) { a[e] = (_Float16)c0[e]; a[e + 8] = (_Float16)c1[e]; }
    afrag[kt] = a;
  }

  const int mode = wave;  // uniform within wave
  const _Float16* pack;
  const float* bias;
  if (mode == 0)      { pack = ws + WQ_PACK; bias = bq; }
  else if (mode == 1) { pack = ws + WK_PACK; bias = bk; }
  else                { pack = ws + WV_PACK; bias = bv; }
  const int nt0 = (mode == 3) ? 8 : 0;

  #pragma unroll 1
  for (int t = 0; t < 4; ++t) {
    const int ntA = nt0 + 2 * t;
    const int ntB = ntA + 1;
    const int nA  = ntA * 16 + lrow;
    const int nB  = ntB * 16 + lrow;
    const float bvA = bias[nA];
    const float bvB = bias[nB];
    const _Float16* ptA = pack + (size_t)ntA * 8 * 512 + (size_t)lane * 16;
    const _Float16* ptB = pack + (size_t)ntB * 8 * 512 + (size_t)lane * 16;
    v8f accA, accB;
    #pragma unroll
    for (int r = 0; r < 8; ++r) { accA[r] = bvA; accB[r] = bvB; }
    v16h bA = *(const v16h*)(ptA);
    v16h bB = *(const v16h*)(ptB);
    #pragma unroll
    for (int kt = 0; kt < 8; ++kt) {
      v16h nxA = bA, nxB = bB;
      if (kt + 1 < 8) {
        nxA = *(const v16h*)(ptA + (size_t)(kt + 1) * 512);
        nxB = *(const v16h*)(ptB + (size_t)(kt + 1) * 512);
      }
      accA = wmma16x16x32(afrag[kt], bA, accA);
      accB = wmma16x16x32(afrag[kt], bB, accB);
      bA = nxA; bB = nxB;
    }

    const int m0 = lhi * 8;  // C/D layout: row M = (lane>>4)*8 + r, col = lane&15
    if (mode == 0) {
      _Float16* q = ws + QH_OFF + (size_t)b * kS * kKS;
      #pragma unroll
      for (int r = 0; r < 8; ++r) {
        q[(size_t)(rtile * 16 + m0 + r) * kKS + nA] = (_Float16)(accA[r] * kScale);
        q[(size_t)(rtile * 16 + m0 + r) * kKS + nB] = (_Float16)(accB[r] * kScale);
      }
    } else if (mode == 1) {
      _Float16* k = ws + KH_OFF + (size_t)b * kS * kKS;
      #pragma unroll
      for (int r = 0; r < 8; ++r) {
        k[(size_t)(rtile * 16 + m0 + r) * kKS + nA] = (_Float16)accA[r];
        k[(size_t)(rtile * 16 + m0 + r) * kKS + nB] = (_Float16)accB[r];
      }
    } else {
      _Float16* vt = ws + VT_OFF + (size_t)b * kVS * kS;
      v8h hvA, hvB;
      #pragma unroll
      for (int r = 0; r < 8; ++r) { hvA[r] = (_Float16)accA[r]; hvB[r] = (_Float16)accB[r]; }
      *(v8h*)(vt + (size_t)nA * kS + rtile * 16 + m0) = hvA;  // contiguous 16B stores
      *(v8h*)(vt + (size_t)nB * kS + rtile * 16 + m0) = hvB;
    }
  }
}

// ---------------------------------------------------------------------------
// Kernel 3: fused causal attention for one (batch, 16-query tile).
// Phase 1: logits = Qs * K^T into LDS (f16), j-tiles in pairs (2 acc chains
// sharing the Q A-fragments). Phase 2: reference-exact masked softmax.
// Phase 3: P * V with paired V column tiles (one LDS A-frag feeds 2 WMMAs).
// ---------------------------------------------------------------------------
__global__ void __launch_bounds__(128) attn_kernel(const _Float16* __restrict__ ws,
                                                   float* __restrict__ out) {
  __shared__ _Float16 sm[16 * kSMS];   // logits/probs, padded stride
  __shared__ float red[128];
  __shared__ float rowstat[16];

  const int b    = blockIdx.x >> 6;
  const int it   = blockIdx.x & 63;
  const int tid  = threadIdx.x;
  const int wave = tid >> 5;
  const int lane = tid & 31;
  const int lrow = lane & 15;
  const int lhi  = lane >> 4;
  const int kb8  = lhi * 16;
  const int jmax = (it + 1) * 16;  // causal: only j-tiles 0..it needed

  const _Float16* qh = ws + QH_OFF + (size_t)b * kS * kKS;
  const _Float16* kh = ws + KH_OFF + (size_t)b * kS * kKS;
  const _Float16* vt = ws + VT_OFF + (size_t)b * kVS * kS;

  // ---- phase 1: logits (scale pre-folded into Q) ----
  v16h qa[4];
  {
    const _Float16* qrow = qh + (size_t)(it * 16 + lrow) * kKS;
    #pragma unroll
    for (int kt = 0; kt < 4; ++kt) {
      v16h a;
      *(v8h*)&a       = *(const v8h*)(qrow + kt * 32 + kb8);
      *((v8h*)&a + 1) = *(const v8h*)(qrow + kt * 32 + kb8 + 8);
      qa[kt] = a;
    }
  }
  for (int jt = wave; jt <= it; jt += 8) {  // pair (jt, jt+4); tail duplicates jt
    const int jt2 = (jt + 4 <= it) ? jt + 4 : jt;   // wave-uniform
    const _Float16* krow0 = kh + (size_t)(jt  * 16 + lrow) * kKS + lhi * 16;
    const _Float16* krow1 = kh + (size_t)(jt2 * 16 + lrow) * kKS + lhi * 16;
    if (jt + 8 <= it)  // prefetch next pair's K rows (global_prefetch_b8)
      __builtin_prefetch(kh + (size_t)((jt + 8) * 16 + lrow) * kKS + lhi * 16, 0, 3);
    v8f acc0 = {}, acc1 = {};
    v16h b0 = *(const v16h*)(krow0);  // contiguous: B column == K row
    v16h b1 = *(const v16h*)(krow1);
    #pragma unroll
    for (int kt = 0; kt < 4; ++kt) {
      v16h n0 = b0, n1 = b1;
      if (kt + 1 < 4) {
        n0 = *(const v16h*)(krow0 + (kt + 1) * 32);
        n1 = *(const v16h*)(krow1 + (kt + 1) * 32);
      }
      acc0 = wmma16x16x32(qa[kt], b0, acc0);
      acc1 = wmma16x16x32(qa[kt], b1, acc1);
      b0 = n0; b1 = n1;
    }
    const int m0 = lhi * 8;
    #pragma unroll
    for (int r = 0; r < 8; ++r)
      sm[(m0 + r) * kSMS + jt * 16 + lrow] = (_Float16)acc0[r];
    if (jt2 != jt) {
      #pragma unroll
      for (int r = 0; r < 8; ++r)
        sm[(m0 + r) * kSMS + jt2 * 16 + lrow] = (_Float16)acc1[r];
    }
  }
  if ((it & 1) == 0) {  // zero-pad the half-filled 32-wide k-tile for phase 3
    for (int idx = tid; idx < 256; idx += 128)
      sm[(idx >> 4) * kSMS + jmax + (idx & 15)] = (_Float16)0.0f;
  }
  __syncthreads();

  // ---- phase 2: masked softmax, 8 threads per row ----
  const int r  = tid & 15;
  const int ch = tid >> 4;            // 0..7
  const int cl = jmax >> 3;           // = 2*(it+1), exact split
  const int j0 = ch * cl;
  const int j1 = j0 + cl;
  const int g  = it * 16 + r;         // strict causal bound: j < g
  _Float16* srow = sm + r * kSMS;

  float mloc = 0.0f;                  // reference uses initial=0.0
  for (int j = j0; j < j1; ++j)
    if (j < g) mloc = fmaxf(mloc, (float)srow[j]);
  red[r * 8 + ch] = mloc;
  __syncthreads();
  if (tid < 16) {
    float m = red[tid * 8];
    for (int c2 = 1; c2 < 8; ++c2) m = fmaxf(m, red[tid * 8 + c2]);
    rowstat[tid] = m;
  }
  __syncthreads();
  const float m = rowstat[r];
  float sloc = 0.0f;
  for (int j = j0; j < j1; ++j) {
    float p = 0.0f;
    if (j < g) { p = __expf((float)srow[j] - m); sloc += p; }
    srow[j] = (_Float16)p;            // exp <= 1, safe in f16
  }
  red[r * 8 + ch] = sloc;
  __syncthreads();
  if (tid < 16) {
    float s2 = 0.0f;
    for (int c2 = 0; c2 < 8; ++c2) s2 += red[tid * 8 + c2];
    rowstat[tid] = 1.0f / (s2 + 1e-16f);
  }
  __syncthreads();
  const float inv = rowstat[r];
  for (int j = j0; j < j1; ++j) srow[j] = (_Float16)((float)srow[j] * inv);
  __syncthreads();

  // ---- phase 3: out = P * V, paired column tiles, V double-buffered ----
  const int kts = (it + 2) >> 1;      // # of 32-wide k-tiles covering jmax
  #pragma unroll 1
  for (int t = 0; t < 2; ++t) {
    const int ntA = wave * 4 + 2 * t;
    const int ntB = ntA + 1;
    v8f accA = {}, accB = {};
    const _Float16* vrowA = vt + (size_t)(ntA * 16 + lrow) * kS + lhi * 16;
    const _Float16* vrowB = vt + (size_t)(ntB * 16 + lrow) * kS + lhi * 16;
    const _Float16* prow  = sm + lrow * kSMS + kb8;
    v16h bA = *(const v16h*)(vrowA);   // contiguous from V^T
    v16h bB = *(const v16h*)(vrowB);
    for (int kt = 0; kt < kts; ++kt) {
      v16h a;
      unsigned int* au = (unsigned int*)&a;
      const _Float16* p = prow + kt * 32;
      #pragma unroll
      for (int q2 = 0; q2 < 4; ++q2) {
        au[q2]     = *(const unsigned int*)(p + 2 * q2);        // K = kb8+{0..7}
        au[q2 + 4] = *(const unsigned int*)(p + 8 + 2 * q2);    // K = kb8+8+{0..7}
      }
      v16h nxA = bA, nxB = bB;
      if (kt + 1 < kts) {
        nxA = *(const v16h*)(vrowA + (kt + 1) * 32);
        nxB = *(const v16h*)(vrowB + (kt + 1) * 32);
      }
      accA = wmma16x16x32(a, bA, accA);   // one LDS A-frag feeds both chains
      accB = wmma16x16x32(a, bB, accB);
      bA = nxA; bB = nxB;
    }
    const int nA = ntA * 16 + lrow;
    const int nB = ntB * 16 + lrow;
    const int m0 = lhi * 8;
    #pragma unroll
    for (int rr = 0; rr < 8; ++rr) {
      out[(size_t)(b * kS + it * 16 + m0 + rr) * kVS + nA] = accA[rr];
      out[(size_t)(b * kS + it * 16 + m0 + rr) * kVS + nB] = accB[rr];
    }
  }
}

extern "C" void kernel_launch(void* const* d_in, const int* in_sizes, int n_in,
                              void* d_out, int out_size, void* d_ws, size_t ws_size,
                              hipStream_t stream) {
  (void)in_sizes; (void)n_in; (void)out_size; (void)ws_size;
  const float* x  = (const float*)d_in[0];
  const float* Wq = (const float*)d_in[1];
  const float* bq = (const float*)d_in[2];
  const float* Wk = (const float*)d_in[3];
  const float* bk = (const float*)d_in[4];
  const float* Wv = (const float*)d_in[5];
  const float* bv = (const float*)d_in[6];
  float* out = (float*)d_out;
  _Float16* ws = (_Float16*)d_ws;  // needs ~33.8 MB

  pack_weights_kernel<<<64, 128, 0, stream>>>(Wq, Wk, Wv, ws);
  qkv_proj_kernel<<<kNB * (kS / 16), 128, 0, stream>>>(x, bq, bk, bv, ws);
  attn_kernel<<<kNB * (kS / 16), 128, 0, stream>>>(ws, out);
}
